// GraphLstmActor_37684043055188
// MI455X (gfx1250) — compile-verified, matrix operand order checked by reading, not run on hardware
//
#include <hip/hip_runtime.h>
#include <hip/hip_bf16.h>
#include <math.h>
#include <stdint.h>

typedef __bf16 bf16;
typedef __bf16 v16bf __attribute__((ext_vector_type(16)));
typedef float  v8f   __attribute__((ext_vector_type(8)));

#define B_TOT   4096
#define NNODE   64
#define CC      16
#define HEADS   4
#define COUT_   32
#define HID     128
#define ADIM    6
#define TS      16                  // samples per workgroup
#define THREADS 256                 // 8 waves (wave32)
#define ROWS    (TS * NNODE)        // 1024 h-rows per WG
#define CHUNK_NODES 4
#define CHUNK_K (CHUNK_NODES * HID) // 512
#define NCHUNK  (NNODE / CHUNK_NODES)

// ---------------- WMMA helpers (CDNA5 v_wmma_f32_16x16x32_bf16) ----------------

__device__ __forceinline__ v8f wmma_bf16(v16bf a, v16bf b, v8f c) {
  // 8 args: (neg_a, A, neg_b, B, c_mod, C, reuse_a, reuse_b)
  return __builtin_amdgcn_wmma_f32_16x16x32_bf16(false, a, false, b, (short)0, c,
                                                 false, false);
}

// A fragment (16x32 bf16) from row-major bf16 LDS buffer.
// Lane L: row = m0 + (L&15); K-half = L>>4.
// v[i], i<8 -> K = k0 + 8*half + i ;  v[8+i] -> K = k0 + 8*half + 16 + i
__device__ __forceinline__ v16bf afrag_lds(const bf16* buf, int stride, int m0,
                                           int k0, int lane) {
  const bf16* p = buf + (m0 + (lane & 15)) * stride + k0 + ((lane >> 4) << 3);
  v16bf v;
#pragma unroll
  for (int i = 0; i < 8; ++i) v[i] = p[i];
#pragma unroll
  for (int i = 0; i < 8; ++i) v[8 + i] = p[16 + i];
  return v;
}

// B fragment (32x16) converted on the fly from f32 row-major weight [K][stride].
// (fallback path when no workspace is available for pre-swizzled weights)
__device__ __forceinline__ v16bf bfrag_f32(const float* W, int stride, int k0,
                                           int n0, int lane) {
  const float* p = W + (size_t)(k0 + ((lane >> 4) << 3)) * stride + n0 + (lane & 15);
  v16bf v;
#pragma unroll
  for (int i = 0; i < 8; ++i) v[i] = (bf16)p[(size_t)i * stride];
#pragma unroll
  for (int i = 0; i < 8; ++i) v[8 + i] = (bf16)p[(size_t)(16 + i) * stride];
  return v;
}

// B fragment from bf16 weights pre-swizzled into fragment order:
// fragment (kt, nt) is 32 lanes x 16 bf16, contiguous 32B per lane ->
// a single aligned v16bf load (two global_load_b128).
__device__ __forceinline__ v16bf bfrag_q(const bf16* Wq, int ntiles, int kt,
                                         int nt, int lane) {
  const bf16* p = Wq + ((size_t)(kt * ntiles + nt) * 32 + lane) * 16;
  return *(const v16bf*)p;
}

// B fragment for a K=16 weight (GAT W is [16][stride]); K>=16 zero-padded.
__device__ __forceinline__ v16bf bfrag_f32_k16(const float* W, int stride, int n0,
                                               int lane) {
  int kh = (lane >> 4) << 3;
  int n  = n0 + (lane & 15);
  v16bf v;
#pragma unroll
  for (int i = 0; i < 8; ++i) v[i] = (bf16)W[(kh + i) * stride + n];
#pragma unroll
  for (int i = 0; i < 8; ++i) v[8 + i] = (bf16)0.0f;
  return v;
}

// A fragment straight from global x (f32), K = 16 channels, zero-padded to 32.
// Row = m0 + (lane&15) maps to (sample = row/64, node = row%64).
__device__ __forceinline__ v16bf afrag_x(const float* x, int s0, int m0, int lane) {
  int row = m0 + (lane & 15);
  int s = row >> 6, n = row & 63;
  int kh = (lane >> 4) << 3;  // channels 0..7 or 8..15
  const float* p = x + (size_t)(s0 + s) * (NNODE * CC) + n * CC + kh;
  v16bf v;
#pragma unroll
  for (int i = 0; i < 8; ++i) v[i] = (bf16)p[i];
#pragma unroll
  for (int i = 0; i < 8; ++i) v[8 + i] = (bf16)0.0f;
  return v;
}

// ---------------- weight pre-swizzle kernel (f32 [K][N] -> bf16 fragments) ----

__global__ __launch_bounds__(256)
void prep_weights(const float* __restrict__ W, bf16* __restrict__ out,
                  int K, int N) {
  int ntiles = N >> 4;
  int total = (K >> 5) * ntiles * 32;          // one thread per (tile, lane)
  int t = blockIdx.x * blockDim.x + threadIdx.x;
  if (t >= total) return;
  int tile = t >> 5, lane = t & 31;
  int kt = tile / ntiles, nt = tile - kt * ntiles;
  int n  = (nt << 4) + (lane & 15);
  int kh = (kt << 5) + ((lane >> 4) << 3);
  bf16* o = out + (size_t)t * 16;
#pragma unroll
  for (int i = 0; i < 8; ++i) o[i]     = (bf16)W[(size_t)(kh + i) * N + n];
#pragma unroll
  for (int i = 0; i < 8; ++i) o[8 + i] = (bf16)W[(size_t)(kh + 16 + i) * N + n];
}

// ---------------- fused network kernel ----------------

__global__ __launch_bounds__(THREADS)
void fused_gat_lstm(const float* __restrict__ x,
                    const float* actor_h, const float* actor_c,
                    const float* critic_h, const float* critic_c,
                    const float* gat1_W, const float* gat1_b, const float* gat1_a,
                    const float* d1_W, const float* d1_b,
                    const float* l1_Wi, const float* l1_Wh, const float* l1_b,
                    const float* ln1_s, const float* ln1_bi,
                    const float* d2_W, const float* d2_b,
                    const float* d3_W, const float* d3_b,
                    const float* dout_W, const float* dout_b,
                    const float* gat2_W, const float* gat2_b, const float* gat2_a,
                    const float* c1_W, const float* c1_b,
                    const float* l2_Wi, const float* l2_Wh, const float* l2_b,
                    const float* ln2_s, const float* ln2_bi,
                    const float* c2_W, const float* c2_b,
                    const float* c3_W, const float* c3_b,
                    const float* coutW, const float* coutb,
                    const bf16* q_d1a, const bf16* q_d1c,
                    const bf16* q_wia, const bf16* q_wha,
                    const bf16* q_wic, const bf16* q_whc,
                    const bf16* q_w2a, const bf16* q_w2c,
                    const bf16* q_w3a, const bf16* q_w3c,
                    float* out_logits, float* out_value,
                    float* out_ah, float* out_ac, float* out_ch, float* out_cc) {
  extern __shared__ char smem[];
  // ---- LDS carve-out (312 KB dynamic) ----
  bf16*  h_lds  = (bf16*)smem;                         // [1024][128] bf16 = 256 KB (stages A-C)
  float* z_f    = (float*)smem;                        // [16][512] f32 = 32 KB (aliases h, stage D+)
  float* nh_f   = (float*)(smem + 32768);              // [16][128] f32 = 8 KB
  float* mu_s   = (float*)(smem + 40960);              // [16]
  float* rv_s   = (float*)(smem + 41088);              // [16]
  bf16*  ln_bf  = (bf16*)(smem + 41216);               // [16][128] bf16 = 4 KB
  bf16*  t2_bf  = (bf16*)(smem + 45312);               // 4 KB
  float* t3_f   = (float*)(smem + 49408);              // 8 KB
  float* ls_lds = (float*)(smem + 262144);             // [1024][4] = 16 KB
  float* ld_lds = (float*)(smem + 278528);             // 16 KB
  bf16*  gchunk = (bf16*)(smem + 294912);              // [16][512] bf16 = 16 KB
  bf16*  amh_bf = (bf16*)(smem + 311296);              // [16][128] bf16 = 4 KB
  bf16*  hp_bf  = (bf16*)(smem + 315392);              // [16][128] bf16 = 4 KB

  const int tid  = threadIdx.x;
  const int lane = tid & 31;
  const int w    = tid >> 5;          // wave id 0..7
  const int s0   = blockIdx.x * TS;   // global sample base

  const float* gW[2]  = {gat1_W, gat2_W};
  const float* gB[2]  = {gat1_b, gat2_b};
  const float* gA[2]  = {gat1_a, gat2_a};
  const float* dW1[2] = {d1_W, c1_W};
  const float* dB1[2] = {d1_b, c1_b};
  const float* Wi[2]  = {l1_Wi, l2_Wi};
  const float* Wh[2]  = {l1_Wh, l2_Wh};
  const float* lb[2]  = {l1_b, l2_b};
  const float* lns[2] = {ln1_s, ln2_s};
  const float* lnb[2] = {ln1_bi, ln2_bi};
  const float* W2[2]  = {d2_W, c2_W};
  const float* B2[2]  = {d2_b, c2_b};
  const float* W3[2]  = {d3_W, c3_W};
  const float* B3[2]  = {d3_b, c3_b};
  const float* hin[2] = {actor_h, critic_h};
  const float* cinp[2] = {actor_c, critic_c};
  float* hout[2] = {out_ah, out_ch};
  float* cout_[2] = {out_ac, out_cc};
  const bf16* qd1[2] = {q_d1a, q_d1c};
  const bf16* qwi[2] = {q_wia, q_wic};
  const bf16* qwh[2] = {q_wha, q_whc};
  const bf16* qw2[2] = {q_w2a, q_w2c};
  const bf16* qw3[2] = {q_w3a, q_w3c};

  for (int br = 0; br < 2; ++br) {
    // ---- Stage A: h = feats @ gatW + gb  -> h_lds (bf16) ----
    {
      v16bf bw = bfrag_f32_k16(gW[br], HID, w * 16, lane);
      int col = w * 16 + (lane & 15);
      int rb  = (lane >> 4) << 3;
      float bias = gB[br][col];
      for (int m0 = 0; m0 < ROWS; m0 += 16) {
        v16bf a = afrag_x(x, s0, m0, lane);
        v8f acc = {};
        acc = wmma_bf16(a, bw, acc);
#pragma unroll
        for (int i = 0; i < 8; ++i)
          h_lds[(m0 + rb + i) * HID + col] = (bf16)(acc[i] + bias);
      }
    }
    __syncthreads();

    // ---- Stage B: per (row, head) source/dest attention logits ----
    for (int t = 0; t < 16; ++t) {
      int idx = tid * 16 + t;            // 0..4095 -> row*4 + head
      int row = idx >> 2, hd = idx & 3;
      const float* av = gA[br] + hd * (2 * COUT_);
      const bf16* hp = h_lds + row * HID + hd * COUT_;
      float ls = 0.f, ld = 0.f;
#pragma unroll
      for (int c = 0; c < COUT_; ++c) {
        float hv = (float)hp[c];
        ls += hv * av[c];
        ld += hv * av[COUT_ + c];
      }
      ls_lds[idx] = ls;
      ld_lds[idx] = ld;
    }
    __syncthreads();

    // ---- Stage C: grid attention + fused d1 GEMM (K = 8192, 16 chunks) ----
    v8f am_acc = {};
    {
      int myS = tid >> 4;            // sample 0..15
      int myNn = (tid >> 2) & 3;     // node-in-chunk
      int myH = tid & 3;             // head
      for (int ch = 0; ch < NCHUNK; ++ch) {
        int n = ch * CHUNK_NODES + myNn;
        int r = n >> 3, c = n & 7;
        int  nbr[5];
        bool vld[5];
        nbr[0] = n;                  vld[0] = true;
        nbr[1] = (r > 0) ? n - 8 : n; vld[1] = (r > 0);
        nbr[2] = (r < 7) ? n + 8 : n; vld[2] = (r < 7);
        nbr[3] = (c > 0) ? n - 1 : n; vld[3] = (c > 0);
        nbr[4] = (c < 7) ? n + 1 : n; vld[4] = (c < 7);
        float lsv = ls_lds[(myS * NNODE + n) * 4 + myH];
        float e[5], emax = -1e30f;
#pragma unroll
        for (int k = 0; k < 5; ++k) {
          float v = lsv + ld_lds[(myS * NNODE + nbr[k]) * 4 + myH];
          v = (v > 0.f) ? v : 0.2f * v;          // leaky_relu(0.2)
          e[k] = vld[k] ? v : -1e30f;
          emax = fmaxf(emax, e[k]);
        }
        float esum = 0.f;
#pragma unroll
        for (int k = 0; k < 5; ++k) { e[k] = __expf(e[k] - emax); esum += e[k]; }
        float rinv = 1.0f / esum;
        float aout[COUT_];
#pragma unroll
        for (int cc = 0; cc < COUT_; ++cc) aout[cc] = 0.f;
#pragma unroll
        for (int k = 0; k < 5; ++k) {
          float aw = e[k] * rinv;
          const bf16* hp = h_lds + (myS * NNODE + nbr[k]) * HID + myH * COUT_;
#pragma unroll
          for (int cc = 0; cc < COUT_; ++cc) aout[cc] += aw * (float)hp[cc];
        }
        bf16* gout = gchunk + myS * CHUNK_K + myNn * HID + myH * COUT_;
#pragma unroll
        for (int cc = 0; cc < COUT_; ++cc) gout[cc] = (bf16)aout[cc];
        __syncthreads();

        // prefetch next chunk of d1 weights (global_prefetch_b8)
        if (ch + 1 < NCHUNK) {
          if (qd1[br])
            __builtin_prefetch(qd1[br] + (size_t)(ch + 1) * CHUNK_K * HID + lane * 64, 0, 1);
          else
            __builtin_prefetch(dW1[br] + (size_t)(ch + 1) * CHUNK_K * HID + lane * 64, 0, 1);
        }

        // each wave owns a 16-col tile of am, accumulate K=512 of this chunk
        for (int kk = 0; kk < CHUNK_K; kk += 32) {
          v16bf a = afrag_lds(gchunk, CHUNK_K, 0, kk, lane);
          v16bf b = qd1[br]
                      ? bfrag_q(qd1[br], HID / 16, (ch * CHUNK_K + kk) >> 5, w, lane)
                      : bfrag_f32(dW1[br], HID, ch * CHUNK_K + kk, w * 16, lane);
          am_acc = wmma_bf16(a, b, am_acc);
        }
        __syncthreads();
      }
    }
    // am = tanh(acc + d1_b) -> bf16; also stage hprev as bf16
    {
      int col = w * 16 + (lane & 15);
      int rb  = (lane >> 4) << 3;
      float bias = dB1[br][col];
#pragma unroll
      for (int i = 0; i < 8; ++i)
        amh_bf[(rb + i) * HID + col] = (bf16)tanhf(am_acc[i] + bias);
    }
    for (int t = 0; t < 8; ++t) {
      int idx = tid + t * THREADS;      // (s, j)
      int s = idx >> 7, j = idx & 127;
      hp_bf[idx] = (bf16)hin[br][(size_t)(s0 + s) * HID + j];
    }
    __syncthreads();

    // ---- Stage D: LSTM preact z = am@Wi + hprev@Wh + b  -> z_f [16][512] ----
    for (int t = 0; t < 4; ++t) {
      int n0 = (w * 4 + t) * 16;
      v8f acc = {};
#pragma unroll
      for (int k0 = 0; k0 < HID; k0 += 32) {
        v16bf aa = afrag_lds(amh_bf, HID, 0, k0, lane);
        v16bf bi = qwi[br] ? bfrag_q(qwi[br], 32, k0 >> 5, n0 >> 4, lane)
                           : bfrag_f32(Wi[br], 4 * HID, k0, n0, lane);
        acc = wmma_bf16(aa, bi, acc);
        v16bf ah = afrag_lds(hp_bf, HID, 0, k0, lane);
        v16bf bh = qwh[br] ? bfrag_q(qwh[br], 32, k0 >> 5, n0 >> 4, lane)
                           : bfrag_f32(Wh[br], 4 * HID, k0, n0, lane);
        acc = wmma_bf16(ah, bh, acc);
      }
      int col = n0 + (lane & 15);
      int rb  = (lane >> 4) << 3;
      float bias = lb[br][col];
#pragma unroll
      for (int i = 0; i < 8; ++i) z_f[(rb + i) * (4 * HID) + col] = acc[i] + bias;
    }
    __syncthreads();

    // ---- Stage E: LSTM gates ----
    for (int t = 0; t < 8; ++t) {
      int idx = tid + t * THREADS;
      int s = idx >> 7, j = idx & 127;
      float zi = z_f[s * 512 + j];
      float zf = z_f[s * 512 + 128 + j];
      float zg = z_f[s * 512 + 256 + j];
      float zo = z_f[s * 512 + 384 + j];
      float cprev = cinp[br][(size_t)(s0 + s) * HID + j];
      float si = 1.f / (1.f + __expf(-zi));
      float sf = 1.f / (1.f + __expf(-zf));
      float so = 1.f / (1.f + __expf(-zo));
      float nc = sf * cprev + si * tanhf(zg);
      float nh = so * tanhf(nc);
      cout_[br][(size_t)(s0 + s) * HID + j] = nc;
      hout[br][(size_t)(s0 + s) * HID + j] = nh;
      nh_f[idx] = nh;
    }
    __syncthreads();

    // ---- Stage F: LayerNorm ----
    if (tid < TS) {
      float m = 0.f;
      for (int j = 0; j < HID; ++j) m += nh_f[tid * HID + j];
      m *= (1.0f / HID);
      float v = 0.f;
      for (int j = 0; j < HID; ++j) { float d = nh_f[tid * HID + j] - m; v += d * d; }
      v *= (1.0f / HID);
      mu_s[tid] = m;
      rv_s[tid] = rsqrtf(v + 1e-6f);
    }
    __syncthreads();
    for (int t = 0; t < 8; ++t) {
      int idx = tid + t * THREADS;
      int s = idx >> 7, j = idx & 127;
      float v = (nh_f[idx] - mu_s[s]) * rv_s[s] * lns[br][j] + lnb[br][j];
      ln_bf[idx] = (bf16)v;
    }
    __syncthreads();

    // ---- Stage G: d2 then d3 (tanh dense 128->128), WMMA per-wave column tile ----
    {
      int n0 = w * 16;
      int col = n0 + (lane & 15);
      int rb  = (lane >> 4) << 3;
      v8f acc = {};
#pragma unroll
      for (int k0 = 0; k0 < HID; k0 += 32) {
        v16bf aa = afrag_lds(ln_bf, HID, 0, k0, lane);
        v16bf bb = qw2[br] ? bfrag_q(qw2[br], 8, k0 >> 5, w, lane)
                           : bfrag_f32(W2[br], HID, k0, n0, lane);
        acc = wmma_bf16(aa, bb, acc);
      }
      float bias = B2[br][col];
#pragma unroll
      for (int i = 0; i < 8; ++i)
        t2_bf[(rb + i) * HID + col] = (bf16)tanhf(acc[i] + bias);
    }
    __syncthreads();
    {
      int n0 = w * 16;
      int col = n0 + (lane & 15);
      int rb  = (lane >> 4) << 3;
      v8f acc = {};
#pragma unroll
      for (int k0 = 0; k0 < HID; k0 += 32) {
        v16bf aa = afrag_lds(t2_bf, HID, 0, k0, lane);
        v16bf bb = qw3[br] ? bfrag_q(qw3[br], 8, k0 >> 5, w, lane)
                           : bfrag_f32(W3[br], HID, k0, n0, lane);
        acc = wmma_bf16(aa, bb, acc);
      }
      float bias = B3[br][col];
#pragma unroll
      for (int i = 0; i < 8; ++i)
        t3_f[(rb + i) * HID + col] = tanhf(acc[i] + bias);
    }
    __syncthreads();

    // ---- Stage H: output heads ----
    if (br == 0) {
      if (tid < TS * ADIM) {
        int s = tid / ADIM, a = tid % ADIM;
        float acc = dout_b[a];
        for (int k = 0; k < HID; ++k) acc += t3_f[s * HID + k] * dout_W[k * ADIM + a];
        out_logits[(size_t)(s0 + s) * ADIM + a] = acc;
      }
    } else {
      if (tid < TS) {
        float acc = coutb[0];
        for (int k = 0; k < HID; ++k) acc += t3_f[tid * HID + k] * coutW[k];
        out_value[s0 + tid] = acc;
      }
    }
    __syncthreads();
  }
}

extern "C" void kernel_launch(void* const* d_in, const int* in_sizes, int n_in,
                              void* d_out, int out_size, void* d_ws, size_t ws_size,
                              hipStream_t stream) {
  (void)in_sizes; (void)n_in; (void)out_size;
  const float* x        = (const float*)d_in[0];
  /* d_in[1] = adj (int32) — static 4-connected grid, neighbors computed in-kernel */
  const float* actor_h  = (const float*)d_in[2];
  const float* actor_c  = (const float*)d_in[3];
  const float* critic_h = (const float*)d_in[4];
  const float* critic_c = (const float*)d_in[5];
  const float* gat1_W = (const float*)d_in[6];
  const float* gat1_b = (const float*)d_in[7];
  const float* gat1_a = (const float*)d_in[8];
  const float* d1_W   = (const float*)d_in[9];
  const float* d1_b   = (const float*)d_in[10];
  const float* l1_Wi  = (const float*)d_in[11];
  const float* l1_Wh  = (const float*)d_in[12];
  const float* l1_b   = (const float*)d_in[13];
  const float* ln1_s  = (const float*)d_in[14];
  const float* ln1_bi = (const float*)d_in[15];
  const float* d2_W   = (const float*)d_in[16];
  const float* d2_b   = (const float*)d_in[17];
  const float* d3_W   = (const float*)d_in[18];
  const float* d3_b   = (const float*)d_in[19];
  const float* dout_W = (const float*)d_in[20];
  const float* dout_b = (const float*)d_in[21];
  const float* gat2_W = (const float*)d_in[22];
  const float* gat2_b = (const float*)d_in[23];
  const float* gat2_a = (const float*)d_in[24];
  const float* c1_W   = (const float*)d_in[25];
  const float* c1_b   = (const float*)d_in[26];
  const float* l2_Wi  = (const float*)d_in[27];
  const float* l2_Wh  = (const float*)d_in[28];
  const float* l2_b   = (const float*)d_in[29];
  const float* ln2_s  = (const float*)d_in[30];
  const float* ln2_bi = (const float*)d_in[31];
  const float* c2_W   = (const float*)d_in[32];
  const float* c2_b   = (const float*)d_in[33];
  const float* c3_W   = (const float*)d_in[34];
  const float* c3_b   = (const float*)d_in[35];
  const float* coutW  = (const float*)d_in[36];
  const float* coutb  = (const float*)d_in[37];

  float* out = (float*)d_out;
  float* out_logits = out;                               // [B,6]
  float* out_value  = out_logits + (size_t)B_TOT * ADIM; // [B]
  float* out_ah     = out_value + B_TOT;                 // [B,128]
  float* out_ac     = out_ah + (size_t)B_TOT * HID;
  float* out_ch     = out_ac + (size_t)B_TOT * HID;
  float* out_cc     = out_ch + (size_t)B_TOT * HID;

  // ---- pre-swizzled bf16 weight fragments in workspace (with fallback) ----
  bf16 *q_d1a = nullptr, *q_d1c = nullptr, *q_wia = nullptr, *q_wha = nullptr,
       *q_wic = nullptr, *q_whc = nullptr, *q_w2a = nullptr, *q_w2c = nullptr,
       *q_w3a = nullptr, *q_w3c = nullptr;
  const size_t SZ_D1 = (size_t)8192 * 128 * 2;   // 2 MB
  const size_t SZ_WL = (size_t)128 * 512 * 2;    // 128 KB
  const size_t SZ_WS = (size_t)128 * 128 * 2;    // 32 KB
  const size_t NEED = 2 * SZ_D1 + 4 * SZ_WL + 4 * SZ_WS;
  if (d_ws && ws_size >= NEED) {
    char* p = (char*)d_ws;
    q_d1a = (bf16*)p;                 p += SZ_D1;
    q_d1c = (bf16*)p;                 p += SZ_D1;
    q_wia = (bf16*)p;                 p += SZ_WL;
    q_wha = (bf16*)p;                 p += SZ_WL;
    q_wic = (bf16*)p;                 p += SZ_WL;
    q_whc = (bf16*)p;                 p += SZ_WL;
    q_w2a = (bf16*)p;                 p += SZ_WS;
    q_w2c = (bf16*)p;                 p += SZ_WS;
    q_w3a = (bf16*)p;                 p += SZ_WS;
    q_w3c = (bf16*)p;                 p += SZ_WS;

    struct { const float* W; bf16* Q; int K, N; } jobs[10] = {
        {d1_W, q_d1a, 8192, 128}, {c1_W, q_d1c, 8192, 128},
        {l1_Wi, q_wia, 128, 512}, {l1_Wh, q_wha, 128, 512},
        {l2_Wi, q_wic, 128, 512}, {l2_Wh, q_whc, 128, 512},
        {d2_W, q_w2a, 128, 128}, {c2_W, q_w2c, 128, 128},
        {d3_W, q_w3a, 128, 128}, {c3_W, q_w3c, 128, 128}};
    for (int j = 0; j < 10; ++j) {
      int total = (jobs[j].K >> 5) * (jobs[j].N >> 4) * 32;
      prep_weights<<<(total + 255) / 256, 256, 0, stream>>>(jobs[j].W, jobs[j].Q,
                                                            jobs[j].K, jobs[j].N);
    }
  }

  const size_t shmem = 319488;  // 312 KB dynamic LDS (<= 320 KB per WGP)
  (void)hipFuncSetAttribute((const void*)fused_gat_lstm,
                            hipFuncAttributeMaxDynamicSharedMemorySize,
                            (int)shmem);

  fused_gat_lstm<<<B_TOT / TS, THREADS, shmem, stream>>>(
      x, actor_h, actor_c, critic_h, critic_c,
      gat1_W, gat1_b, gat1_a, d1_W, d1_b,
      l1_Wi, l1_Wh, l1_b, ln1_s, ln1_bi,
      d2_W, d2_b, d3_W, d3_b, dout_W, dout_b,
      gat2_W, gat2_b, gat2_a, c1_W, c1_b,
      l2_Wi, l2_Wh, l2_b, ln2_s, ln2_bi,
      c2_W, c2_b, c3_W, c3_b, coutW, coutb,
      q_d1a, q_d1c, q_wia, q_wha, q_wic, q_whc, q_w2a, q_w2c, q_w3a, q_w3c,
      out_logits, out_value, out_ah, out_ac, out_ch, out_cc);
}
